// MonNet_25941602468480
// MI455X (gfx1250) — matrix operand order repaired and checked by reading, not compiled, not run on hardware
//
#include <hip/hip_runtime.h>

// ---------------------------------------------------------------------------
// MonNet fixed point on MI455X (gfx1250, wave32, WMMA bf16 -> f32 accum)
//
//   Ux = x @ U_w^T + U_b                (once)
//   Gt = [(B_w^T - B_w) - A_w^T A_w]^T  (once, stored [N,K] so the hot loop
//                                        needs no LDS transpose)
//   repeat 50x:  z = relu(z @ G + Ux)   (first step with z=0 is relu(Ux))
//
// Hot loop GEMM: bf16 WMMA, f32 accum, double-buffered LDS fed by
// global_load_async_to_lds_b128 (ASYNCcnt), 8 WMMAs per barrier pair.
// ---------------------------------------------------------------------------

#ifndef USE_ASYNC
#define USE_ASYNC 1
#endif

typedef __attribute__((ext_vector_type(16))) __bf16         v16bf;
typedef __attribute__((ext_vector_type(8)))  float          v8f;
typedef __attribute__((ext_vector_type(8)))  unsigned short v8u;

__device__ __forceinline__ unsigned short f32_to_bf16_rne(float f) {
  unsigned int u = __builtin_bit_cast(unsigned int, f);
  u += 0x7FFFu + ((u >> 16) & 1u);            // round-to-nearest-even
  return (unsigned short)(u >> 16);
}

// ------------------------- elementwise helpers -----------------------------

__global__ void k_f32_to_bf16(const float* __restrict__ s,
                              unsigned short* __restrict__ d, int n) {
  int i = blockIdx.x * blockDim.x + threadIdx.x;
  if (i < n) d[i] = f32_to_bf16_rne(s[i]);
}

__global__ void k_transpose_to_bf16(const float* __restrict__ s,
                                    unsigned short* __restrict__ d, int dim) {
  int i = blockIdx.x * blockDim.x + threadIdx.x;
  if (i < dim * dim) {
    int r = i / dim, c = i - r * dim;
    d[i] = f32_to_bf16_rne(s[c * dim + r]);   // d = s^T
  }
}

__global__ void k_skew(const float* __restrict__ s,        // S = s^T - s
                       float* __restrict__ d, int dim) {
  int i = blockIdx.x * blockDim.x + threadIdx.x;
  if (i < dim * dim) {
    int r = i / dim, c = i - r * dim;
    d[i] = s[c * dim + r] - s[i];
  }
}

__global__ void k_relu_to_bf16(const float* __restrict__ s,
                               unsigned short* __restrict__ d, int n) {
  int i = blockIdx.x * blockDim.x + threadIdx.x;
  if (i < n) {
    float v = s[i];
    v = v > 0.0f ? v : 0.0f;
    d[i] = f32_to_bf16_rne(v);
  }
}

// ------------------------- tiled WMMA GEMM ---------------------------------
// acc = A[MxK](bf16) @ B (bf16, stored [N,K] row-major -> direct LDS copy)
//   MODE 0: outF[row*N+col] = acc + aux[col]            (Ux = x U^T + b)
//   MODE 1: outB[col*M+row] = bf16(aux[row*N+col]-acc)  (Gt = (S - A^T A)^T)
//   MODE 2: t = relu(acc + aux[idx]); outB[idx]=bf16(t); if(outF) outF[idx]=t

#define BM 128
#define BN 64
#define BK 64
#define BKP (BK + 8)     // 144B row stride -> conflict-free fragment loads

union Frag { v16bf v; v8u h[2]; };

#if USE_ASYNC
__device__ __forceinline__ void async_b128(unsigned lds_off, const void* g) {
  asm volatile("global_load_async_to_lds_b128 %0, %1, off"
               :: "v"(lds_off), "v"(g) : "memory");
}
__device__ __forceinline__ void async_wait(int pending6) {
  if (pending6) asm volatile("s_wait_asynccnt 0x6" ::: "memory");
  else          asm volatile("s_wait_asynccnt 0x0" ::: "memory");
}
#endif

template <int MODE>
__global__ __launch_bounds__(256)
void gemm_wmma(const unsigned short* __restrict__ A,
               const unsigned short* __restrict__ Bm,   // [N,K] row-major
               const float* __restrict__ aux,
               float* __restrict__ outF,
               unsigned short* __restrict__ outB,
               int M, int N, int K, int ldb) {
  __shared__ unsigned short As[2][BM][BKP];   // [buf][m][k]
  __shared__ unsigned short Bs[2][BN][BKP];   // [buf][n][k]

  const int tid  = threadIdx.x;
  const int lane = tid & 31;
  const int wave = tid >> 5;     // 8 waves
  const int wr   = wave >> 1;    // 0..3  wave row (32 rows each)
  const int wc   = wave & 1;     // 0..1  wave col (32 cols each)
  const int half = lane >> 4;
  const int lm   = lane & 15;

  const int m0 = blockIdx.y * BM;
  const int n0 = blockIdx.x * BN;

  v8f acc[2][2];
  v8f zro = {};
  acc[0][0] = zro; acc[0][1] = zro; acc[1][0] = zro; acc[1][1] = zro;

  // ---- tile fetch: 6x b128 per thread (4 for A 128x64, 2 for B 64x64) ----
  auto issue_tile = [&](int kt, int b) {
    const int kbase = kt * BK;
#pragma unroll
    for (int i = 0; i < 4; ++i) {
      int id = tid + i * 256;
      int r = id >> 3, c8 = (id & 7) * 8;
      const unsigned short* gp = A + (size_t)(m0 + r) * K + kbase + c8;
#if USE_ASYNC
      async_b128((unsigned)(size_t)&As[b][r][c8], gp);
#else
      *(v8u*)&As[b][r][c8] = *(const v8u*)gp;
#endif
    }
#pragma unroll
    for (int i = 0; i < 2; ++i) {
      int id = tid + i * 256;
      int r = id >> 3, c8 = (id & 7) * 8;
      const unsigned short* gp = Bm + (size_t)(n0 + r) * ldb + kbase + c8;
#if USE_ASYNC
      async_b128((unsigned)(size_t)&Bs[b][r][c8], gp);
#else
      *(v8u*)&Bs[b][r][c8] = *(const v8u*)gp;
#endif
    }
  };

  const int ksteps = K / BK;
  issue_tile(0, 0);

  for (int kt = 0; kt < ksteps; ++kt) {
    const int cb = kt & 1;
    // prefetch next tile into the other buffer (its last readers passed the
    // end-of-step barrier of kt-1), then retire this tile's 6 async loads
    if (kt + 1 < ksteps) {
      issue_tile(kt + 1, cb ^ 1);
#if USE_ASYNC
      async_wait(1);
#endif
    } else {
#if USE_ASYNC
      async_wait(0);
#endif
    }
    __syncthreads();

    // ---- 2 K-chunks of 32 -> 8x v_wmma_f32_16x16x32_bf16 per wave ----
#pragma unroll
    for (int kc = 0; kc < 2; ++kc) {
      const int kb = kc * 32;
      const int k1 = kb + half * 8;    // A: lanes<16 K0..7/16..23; >=16 +8
      const int k2 = kb + half * 8;    // B: lanes<16 K0..7/16..23; >=16 +8
      Frag aF[2], bF[2];
#pragma unroll
      for (int t = 0; t < 2; ++t) {
        int m = wr * 32 + t * 16 + lm;
        aF[t].h[0] = *(const v8u*)&As[cb][m][k1];
        aF[t].h[1] = *(const v8u*)&As[cb][m][k1 + 16];
        int n = wc * 32 + t * 16 + lm;
        bF[t].h[0] = *(const v8u*)&Bs[cb][n][k2];
        bF[t].h[1] = *(const v8u*)&Bs[cb][n][k2 + 16];
      }
#pragma unroll
      for (int ti = 0; ti < 2; ++ti)
#pragma unroll
        for (int tj = 0; tj < 2; ++tj)
          acc[ti][tj] = __builtin_amdgcn_wmma_f32_16x16x32_bf16(
              false, aF[ti].v, false, bF[tj].v, (short)0, acc[ti][tj],
              false, false);
    }
    __syncthreads();
  }

  // ---- epilogue (C layout: VGPR r -> row r + 8*half, col = lane&15) ----
#pragma unroll
  for (int ti = 0; ti < 2; ++ti) {
#pragma unroll
    for (int tj = 0; tj < 2; ++tj) {
      int rowBase = m0 + wr * 32 + ti * 16 + half * 8;
      int col     = n0 + wc * 32 + tj * 16 + lm;
#pragma unroll
      for (int r = 0; r < 8; ++r) {
        int row = rowBase + r;
        int idx = row * N + col;
        float v = acc[ti][tj][r];
        if (MODE == 0) {
          outF[idx] = v + aux[col];                      // + bias
        } else if (MODE == 1) {
          outB[col * M + row] = f32_to_bf16_rne(aux[idx] - v);  // store G^T
        } else {
          float t = v + aux[idx];                        // + Ux
          t = t > 0.0f ? t : 0.0f;                       // relu
          outB[idx] = f32_to_bf16_rne(t);
          if (outF) outF[idx] = t;                       // final f32 output
        }
      }
    }
  }
}

// ------------------------------ driver -------------------------------------

extern "C" void kernel_launch(void* const* d_in, const int* in_sizes, int n_in,
                              void* d_out, int out_size, void* d_ws,
                              size_t ws_size, hipStream_t stream) {
  (void)in_sizes; (void)n_in; (void)out_size; (void)ws_size;
  constexpr int BATCH = 2048, IN = 512, OUT = 1024;

  const float* x   = (const float*)d_in[0];  // [2048,512]
  const float* U_w = (const float*)d_in[1];  // [1024,512]
  const float* U_b = (const float*)d_in[2];  // [1024]
  const float* A_w = (const float*)d_in[3];  // [1024,1024]
  const float* B_w = (const float*)d_in[4];  // [1024,1024]

  // workspace layout (all 256B-aligned by construction)
  char* p = (char*)d_ws;
  float*          Ux = (float*)p;          p += (size_t)BATCH * OUT * 4;  // 8 MB
  unsigned short* Gt = (unsigned short*)p; p += (size_t)OUT * OUT * 2;    // 2 MB
  unsigned short* zA = (unsigned short*)p; p += (size_t)BATCH * OUT * 2;  // 4 MB
  unsigned short* zB = (unsigned short*)p; p += (size_t)BATCH * OUT * 2;  // 4 MB
  char* scratch = p;                                                      // 6 MB
  // stage 1 scratch
  unsigned short* xb  = (unsigned short*)scratch;                 // 2 MB
  unsigned short* Uwb = (unsigned short*)(scratch + (2u << 20));  // 1 MB
  // stage 2 scratch (reuses same region)
  unsigned short* AwTb = (unsigned short*)scratch;                // 2 MB
  float*          Skw  = (float*)(scratch + (2u << 20));          // 4 MB

  dim3 blk(256);
  auto eg = [](int n) { return dim3((n + 255) / 256); };

  // ---- stage 1: Ux = x @ U_w^T + U_b  (U_w is already [N,K]) ----
  k_f32_to_bf16<<<eg(BATCH * IN), blk, 0, stream>>>(x, xb, BATCH * IN);
  k_f32_to_bf16<<<eg(OUT * IN), blk, 0, stream>>>(U_w, Uwb, OUT * IN);
  gemm_wmma<0><<<dim3(OUT / BN, BATCH / BM), blk, 0, stream>>>(
      xb, Uwb, U_b, Ux, nullptr, BATCH, OUT, IN, IN);

  // ---- stage 2: Gt = [(B_w^T - B_w) - A_w^T A_w]^T ----
  // acc = AwT @ A_w ; B operand in [N,K] form is exactly AwT, so one copy.
  k_transpose_to_bf16<<<eg(OUT * OUT), blk, 0, stream>>>(A_w, AwTb, OUT);
  k_skew<<<eg(OUT * OUT), blk, 0, stream>>>(B_w, Skw, OUT);
  gemm_wmma<1><<<dim3(OUT / BN, OUT / BM), blk, 0, stream>>>(
      AwTb, AwTb, Skw, nullptr, Gt, OUT, OUT, OUT, OUT);

  // ---- stage 3: z1 = relu(Ux), then 49x  z = relu(z @ G + Ux) ----
  k_relu_to_bf16<<<eg(BATCH * OUT), blk, 0, stream>>>(Ux, zA, BATCH * OUT);

  unsigned short* cur = zA;
  unsigned short* nxt = zB;
  for (int it = 0; it < 49; ++it) {
    float* of = (it == 48) ? (float*)d_out : nullptr;
    gemm_wmma<2><<<dim3(OUT / BN, BATCH / BM), blk, 0, stream>>>(
        cur, Gt, Ux, of, nxt, BATCH, OUT, OUT, OUT);
    unsigned short* t = cur; cur = nxt; nxt = t;
  }
}
